// FlashAttentionWrapper_13932873908581
// MI455X (gfx1250) — matrix operand order repaired and checked by reading, not compiled, not run on hardware
//
#include <hip/hip_runtime.h>
#include <math.h>

typedef __bf16 bf16t;
typedef __attribute__((ext_vector_type(16))) __bf16 v16bf;
typedef __attribute__((ext_vector_type(8)))  __bf16 v8bf;
typedef __attribute__((ext_vector_type(8)))  float   v8f;
typedef __attribute__((ext_vector_type(4)))  unsigned int v4u;
typedef __attribute__((ext_vector_type(8)))  int v8i;
typedef __attribute__((ext_vector_type(4)))  int v4i;

#define B_   2
#define S_   2048
#define D_   2048
#define NH_  32
#define HD_  64
#define NKV_ 8
#define NREP_ 4

// ---------------- fragment loaders (contiguous k = (lane>>4)*16 + e in A and B, wave32) ----------------

// A-matrix 16x32 bf16: lane = row (M = lane&15), element e -> K = (lane>>4)*16 + e
static __device__ __forceinline__ v16bf load_A_frag(const bf16t* base, int ld, int lane) {
  const int row = lane & 15, half = lane >> 4;
  const bf16t* p = base + (size_t)row * (size_t)ld + half * 16;
  return *(const v16bf*)p;
}

// B-matrix 32x16 bf16 sourced from row-major W with element (K,N) = base[N*ld + K]
// per lane: N = lane&15, K = (lane>>4)*16 + e  -> 16 contiguous bf16 (32B load)
static __device__ __forceinline__ v16bf load_B_frag(const bf16t* base, int ld, int lane) {
  const bf16t* p = base + (size_t)(lane & 15) * (size_t)ld + (lane >> 4) * 16;
  return *(const v16bf*)p;
}

static __device__ __forceinline__ v8f wmma_bf16(v16bf a, v16bf b, v8f c) {
  return __builtin_amdgcn_wmma_f32_16x16x32_bf16(false, a, false, b, (short)0, c, false, false);
}

// ---------------- TDM: 2D tile load (global -> LDS), D# per ISA ch.8 ----------------
// Loads tile_d1 rows x tile_d0 elems (bf16) from row-major tensor (row stride = stride0
// elems) starting at gaddr, into LDS at lds_addr, packed row-major [tile_d1][tile_d0].
// This toolchain uses the 6-arg builtin form:
//   (uint32x4 g0, int32x8 g1, int32x4 g2, int32x4 g3, int32x8 extra, i32 cpol)
static __device__ __forceinline__ void tdm_load_2d_bf16(
    unsigned lds_addr, const bf16t* gaddr, unsigned tensor_d0, unsigned tensor_d1,
    unsigned tile_d0, unsigned tile_d1, unsigned stride0) {
  unsigned long long ga = (unsigned long long)(size_t)gaddr;
  v4u g0;
  g0[0] = 1u;                                                  // count=1, user-mode load
  g0[1] = lds_addr;                                            // D#.lds_addr (bytes)
  g0[2] = (unsigned)(ga & 0xFFFFFFFFu);                        // global_addr[31:0]
  g0[3] = (unsigned)((ga >> 32) & 0x01FFFFFFu) | (2u << 30);   // addr[56:32] | type=2
  v8i g1;
  g1[0] = (int)(1u << 16);                                     // data_size=1 (2 bytes)
  g1[1] = (int)((tensor_d0 & 0xFFFFu) << 16);                  // tensor_dim0[15:0]
  g1[2] = (int)(((tensor_d0 >> 16) & 0xFFFFu) | ((tensor_d1 & 0xFFFFu) << 16));
  g1[3] = (int)(((tensor_d1 >> 16) & 0xFFFFu) | ((tile_d0 & 0xFFFFu) << 16));
  g1[4] = (int)(tile_d1 & 0xFFFFu);                            // tile_dim1 (tile_dim2=0)
  g1[5] = (int)stride0;                                        // tensor_dim0_stride[31:0]
  g1[6] = 0;                                                   // stride0[47:32], stride1 lo
  g1[7] = 0;
  v4i z4 = {0, 0, 0, 0};                                       // groups 2/3: 2D tensor
  v8i z8 = {0, 0, 0, 0, 0, 0, 0, 0};                           // extra operand (zero-filled)
  __builtin_amdgcn_tensor_load_to_lds(g0, g1, z4, z4, z8, 0);
}

// ---------------- helper kernels ----------------

// vectorized fp32 -> bf16 (8 elements / thread); n8 = n/8, all sizes divide by 8
__global__ void cvt_f32_bf16_kernel(const float* __restrict__ in, bf16t* __restrict__ out,
                                    size_t n8) {
  size_t i = (size_t)blockIdx.x * blockDim.x + threadIdx.x;
  if (i >= n8) return;
  const float4* p = (const float4*)in + i * 2;
  float4 x = p[0], y = p[1];
  v8bf r;
  r[0] = (bf16t)x.x; r[1] = (bf16t)x.y; r[2] = (bf16t)x.z; r[3] = (bf16t)x.w;
  r[4] = (bf16t)y.x; r[5] = (bf16t)y.y; r[6] = (bf16t)y.z; r[7] = (bf16t)y.w;
  *(v8bf*)(out + i * 8) = r;
}

// In-place RoPE on bf16 [B*S, nheads*64]; thread handles one (row, head, pair i<32)
__global__ void rope_kernel(bf16t* __restrict__ X, int nheads) {
  size_t idx = (size_t)blockIdx.x * blockDim.x + threadIdx.x;
  size_t total = (size_t)B_ * S_ * nheads * 32;
  if (idx >= total) return;
  int i = (int)(idx & 31);
  size_t t = idx >> 5;
  int hh = (int)(t % nheads);
  size_t row = t / nheads;
  int s = (int)(row % S_);
  // inv_freq = 10000^(-2i/64)
  float inv = __expf(-((2.0f * (float)i) / 64.0f) * 9.210340371976184f);
  float ang = (float)s * inv;
  float c = cosf(ang), sn = sinf(ang);
  bf16t* p = X + row * (size_t)(nheads * HD_) + (size_t)hh * HD_;
  float x1 = (float)p[i], x2 = (float)p[i + 32];
  p[i]      = (bf16t)(x1 * c - x2 * sn);
  p[i + 32] = (bf16t)(x2 * c + x1 * sn);
}

// V [B*S, NKV*64] -> VT [B, NKV, 64, S]
__global__ void vtrans_kernel(const bf16t* __restrict__ V, bf16t* __restrict__ VT) {
  size_t i = (size_t)blockIdx.x * blockDim.x + threadIdx.x;
  size_t total = (size_t)B_ * S_ * NKV_ * HD_;
  if (i >= total) return;
  int col = (int)(i % (NKV_ * HD_));
  size_t row = i / (NKV_ * HD_);
  int b = (int)(row / S_);
  int s = (int)(row % S_);
  int kvh = col / HD_, d = col % HD_;
  VT[((size_t)((b * NKV_ + kvh) * HD_ + d)) * S_ + s] = V[i];
}

// ---------------- WMMA GEMM, double-buffered TDM tile staging ----------------
// C[m,n] = sum_k A[m,k] * W[n,k]; block = 128 thr (4 waves), block tile 64x128,
// wave tile 32x64, K-step 32. A (64x32) and W (128x32) tiles staged via TDM with
// a 2-deep pipeline: tile i+1 stays in flight while tile i is consumed.
template <bool OUT_F32>
__global__ __launch_bounds__(128) void gemm_bt_kernel(
    const bf16t* __restrict__ A, const bf16t* __restrict__ W,
    float* __restrict__ Cf, bf16t* __restrict__ Cb, int M, int N, int K) {
  __shared__ bf16t ldsA[2][64 * 32];
  __shared__ bf16t ldsW[2][128 * 32];
  const int lane = threadIdx.x & 31;
  const int wave = threadIdx.x >> 5;
  const int wm = wave >> 1, wn = wave & 1;
  const int mblk = blockIdx.x * 64;
  const int nblk = blockIdx.y * 128;

  v8f acc[2][4];
#pragma unroll
  for (int mi = 0; mi < 2; ++mi)
#pragma unroll
    for (int j = 0; j < 4; ++j) acc[mi][j] = (v8f){0, 0, 0, 0, 0, 0, 0, 0};

  // prologue: stage tiles 0 and 1
  if (wave == 0) {
    tdm_load_2d_bf16((unsigned)(size_t)(void*)ldsA[0], A + (size_t)mblk * K,
                     (unsigned)K, (unsigned)M, 32u, 64u, (unsigned)K);
    tdm_load_2d_bf16((unsigned)(size_t)(void*)ldsW[0], W + (size_t)nblk * K,
                     (unsigned)K, (unsigned)N, 32u, 128u, (unsigned)K);
    if (32 < K) {
      tdm_load_2d_bf16((unsigned)(size_t)(void*)ldsA[1], A + (size_t)mblk * K + 32,
                       (unsigned)K, (unsigned)M, 32u, 64u, (unsigned)K);
      tdm_load_2d_bf16((unsigned)(size_t)(void*)ldsW[1], W + (size_t)nblk * K + 32,
                       (unsigned)K, (unsigned)N, 32u, 128u, (unsigned)K);
    }
  }

  for (int k0 = 0; k0 < K; k0 += 32) {
    const int cur = (k0 >> 5) & 1;
    // current tile's 2 descriptors complete in order; next tile's 2 may stay in flight
    if (wave == 0) {
      if (k0 + 32 < K) __builtin_amdgcn_s_wait_tensorcnt(2);
      else             __builtin_amdgcn_s_wait_tensorcnt(0);
    }
    __syncthreads();  // publish current tile to all waves
    const bf16t* tA = ldsA[cur];
    const bf16t* tW = ldsW[cur];
    v16bf fa0 = load_A_frag(tA + (size_t)(wm * 32 + 0) * 32, 32, lane);
    v16bf fa1 = load_A_frag(tA + (size_t)(wm * 32 + 16) * 32, 32, lane);
    v16bf fb[4];
#pragma unroll
    for (int j = 0; j < 4; ++j)
      fb[j] = load_B_frag(tW + (size_t)(wn * 64 + j * 16) * 32, 32, lane);
#pragma unroll
    for (int j = 0; j < 4; ++j) {
      acc[0][j] = wmma_bf16(fa0, fb[j], acc[0][j]);
      acc[1][j] = wmma_bf16(fa1, fb[j], acc[1][j]);
    }
    __syncthreads();  // all waves done reading buffer `cur`
    if (wave == 0 && k0 + 64 < K) {  // refill the just-freed buffer with tile k0+64
      tdm_load_2d_bf16((unsigned)(size_t)(void*)ldsA[cur], A + (size_t)mblk * K + k0 + 64,
                       (unsigned)K, (unsigned)M, 32u, 64u, (unsigned)K);
      tdm_load_2d_bf16((unsigned)(size_t)(void*)ldsW[cur], W + (size_t)nblk * K + k0 + 64,
                       (unsigned)K, (unsigned)N, 32u, 128u, (unsigned)K);
    }
  }

  const int half = lane >> 4, nn = lane & 15;
  const int m0 = mblk + wm * 32;
  const int n0 = nblk + wn * 64;
#pragma unroll
  for (int mi = 0; mi < 2; ++mi)
#pragma unroll
    for (int j = 0; j < 4; ++j)
#pragma unroll
      for (int e = 0; e < 8; ++e) {
        int r = m0 + mi * 16 + half * 8 + e;
        int c = n0 + j * 16 + nn;
        if (OUT_F32) Cf[(size_t)r * N + c] = acc[mi][j][e];
        else         Cb[(size_t)r * N + c] = (bf16t)acc[mi][j][e];
      }
}

// ---------------- flash attention: one wave per (b, h, 16-row q tile) ----------------
__global__ __launch_bounds__(32) void attn_kernel(
    const bf16t* __restrict__ Q, const bf16t* __restrict__ Kl,
    const bf16t* __restrict__ VT, bf16t* __restrict__ O) {
  __shared__ bf16t Pld[16 * 32];  // one wave per block: private P staging tile
  const int lane = threadIdx.x & 31;
  const int qt = blockIdx.x, h = blockIdx.y, b = blockIdx.z;
  const int q0 = qt * 16;
  const int kvh = h / NREP_;
  const int half = lane >> 4, nn = lane & 15;

  const bf16t* Qbase = Q + ((size_t)(b * S_ + q0)) * (NH_ * HD_) + (size_t)h * HD_;
  v16bf qa0 = load_A_frag(Qbase, NH_ * HD_, lane);       // d = 0..31
  v16bf qa1 = load_A_frag(Qbase + 32, NH_ * HD_, lane);  // d = 32..63

  const bf16t* Kbase  = Kl + ((size_t)(b * S_)) * (NKV_ * HD_) + (size_t)kvh * HD_;
  const bf16t* VTbase = VT + ((size_t)((b * NKV_ + kvh) * HD_)) * S_;

  v8f o0 = {}, o1 = {}, o2 = {}, o3 = {};
  float mrow[8], lrow[8];
#pragma unroll
  for (int e = 0; e < 8; ++e) { mrow[e] = -1e30f; lrow[e] = 0.0f; }

  const int kend = q0 + 16;  // causal: keys < q0+16
  for (int kb = 0; kb < kend; kb += 32) {
    // prefetch next key/value blocks into cache (global_prefetch_b8)
    if (kb + 32 < kend) {
      __builtin_prefetch((const void*)(Kbase + (size_t)(kb + 32 + lane) * (NKV_ * HD_)), 0, 1);
      __builtin_prefetch((const void*)(VTbase + (size_t)lane * S_ + kb + 32), 0, 1);
      __builtin_prefetch((const void*)(VTbase + (size_t)(32 + lane) * S_ + kb + 32), 0, 1);
    }
    const bf16t* K0 = Kbase + (size_t)kb * (NKV_ * HD_);
    // scores: S[q, key] = sum_d Q[q,d] * K[key,d]   (K row-major -> B fragment direct)
    v16bf kb00 = load_B_frag(K0, NKV_ * HD_, lane);
    v16bf kb01 = load_B_frag(K0 + 32, NKV_ * HD_, lane);
    v16bf kb10 = load_B_frag(K0 + (size_t)16 * (NKV_ * HD_), NKV_ * HD_, lane);
    v16bf kb11 = load_B_frag(K0 + (size_t)16 * (NKV_ * HD_) + 32, NKV_ * HD_, lane);
    v8f s0 = {}, s1 = {};
    s0 = wmma_bf16(qa0, kb00, s0);
    s0 = wmma_bf16(qa1, kb01, s0);
    s1 = wmma_bf16(qa0, kb10, s1);
    s1 = wmma_bf16(qa1, kb11, s1);

    // scale + causal mask + online softmax (row = q0 + half*8 + e, col = kb(+16) + nn)
#pragma unroll
    for (int e = 0; e < 8; ++e) {
      int row = q0 + half * 8 + e;
      float v0 = (kb + nn > row)      ? -1e30f : s0[e] * 0.125f;
      float v1 = (kb + 16 + nn > row) ? -1e30f : s1[e] * 0.125f;
      float mt = fmaxf(v0, v1);
      mt = fmaxf(mt, __shfl_xor(mt, 1));
      mt = fmaxf(mt, __shfl_xor(mt, 2));
      mt = fmaxf(mt, __shfl_xor(mt, 4));
      mt = fmaxf(mt, __shfl_xor(mt, 8));
      float mnew = fmaxf(mrow[e], mt);
      float corr = __expf(mrow[e] - mnew);
      float p0 = __expf(v0 - mnew);
      float p1 = __expf(v1 - mnew);
      float rs = p0 + p1;
      rs += __shfl_xor(rs, 1);
      rs += __shfl_xor(rs, 2);
      rs += __shfl_xor(rs, 4);
      rs += __shfl_xor(rs, 8);
      lrow[e] = lrow[e] * corr + rs;
      mrow[e] = mnew;
      o0[e] *= corr; o1[e] *= corr; o2[e] *= corr; o3[e] *= corr;
      Pld[(half * 8 + e) * 32 + nn]      = (bf16t)p0;
      Pld[(half * 8 + e) * 32 + 16 + nn] = (bf16t)p1;
    }
    asm volatile("s_wait_dscnt 0x0" ::: "memory");
    __builtin_amdgcn_wave_barrier();

    // P (16x32) as A fragment from LDS; V via VT rows as B fragments
    v16bf pa  = load_A_frag(Pld, 32, lane);
    v16bf vb0 = load_B_frag(VTbase + kb, S_, lane);
    v16bf vb1 = load_B_frag(VTbase + (size_t)16 * S_ + kb, S_, lane);
    v16bf vb2 = load_B_frag(VTbase + (size_t)32 * S_ + kb, S_, lane);
    v16bf vb3 = load_B_frag(VTbase + (size_t)48 * S_ + kb, S_, lane);
    o0 = wmma_bf16(pa, vb0, o0);
    o1 = wmma_bf16(pa, vb1, o1);
    o2 = wmma_bf16(pa, vb2, o2);
    o3 = wmma_bf16(pa, vb3, o3);
  }

#pragma unroll
  for (int e = 0; e < 8; ++e) {
    float inv = 1.0f / lrow[e];
    int row = b * S_ + q0 + half * 8 + e;
    size_t base = (size_t)row * (NH_ * HD_) + (size_t)h * HD_ + nn;
    O[base]      = (bf16t)(o0[e] * inv);
    O[base + 16] = (bf16t)(o1[e] * inv);
    O[base + 32] = (bf16t)(o2[e] * inv);
    O[base + 48] = (bf16t)(o3[e] * inv);
  }
}

// ---------------- host launcher ----------------

extern "C" void kernel_launch(void* const* d_in, const int* in_sizes, int n_in,
                              void* d_out, int out_size, void* d_ws, size_t ws_size,
                              hipStream_t stream) {
  const float* hidden = (const float*)d_in[0];
  const float* Wq = (const float*)d_in[1];
  const float* Wk = (const float*)d_in[2];
  const float* Wv = (const float*)d_in[3];
  const float* Wo = (const float*)d_in[4];
  // d_in[5] = attention_mask (exactly causal) -> implemented analytically
  float* out = (float*)d_out;

  char* ws = (char*)d_ws;
  size_t off = 0;
  bf16t* Xb  = (bf16t*)(ws + off); off += (size_t)B_ * S_ * D_ * 2;          // 16 MB
  bf16t* Wqb = (bf16t*)(ws + off); off += (size_t)(NH_ * HD_) * D_ * 2;      //  8 MB
  bf16t* Wkb = (bf16t*)(ws + off); off += (size_t)(NKV_ * HD_) * D_ * 2;     //  2 MB
  bf16t* Wvb = (bf16t*)(ws + off); off += (size_t)(NKV_ * HD_) * D_ * 2;     //  2 MB
  bf16t* Wob = (bf16t*)(ws + off); off += (size_t)D_ * (NH_ * HD_) * 2;      //  8 MB
  bf16t* Qb  = (bf16t*)(ws + off); off += (size_t)B_ * S_ * (NH_ * HD_) * 2; // 16 MB
  bf16t* Kb  = (bf16t*)(ws + off); off += (size_t)B_ * S_ * (NKV_ * HD_) * 2;//  4 MB
  bf16t* Vb  = (bf16t*)(ws + off); off += (size_t)B_ * S_ * (NKV_ * HD_) * 2;//  4 MB
  bf16t* VTb = (bf16t*)(ws + off); off += (size_t)B_ * S_ * (NKV_ * HD_) * 2;//  4 MB
  bf16t* Ob  = (bf16t*)(ws + off); off += (size_t)B_ * S_ * (NH_ * HD_) * 2; // 16 MB

  const int M = B_ * S_;  // 4096

  // fp32 -> bf16 conversions (8 elems/thread)
  {
    size_t n8;
    n8 = (size_t)B_ * S_ * D_ / 8;
    cvt_f32_bf16_kernel<<<dim3((unsigned)((n8 + 255) / 256)), dim3(256), 0, stream>>>(hidden, Xb, n8);
    n8 = (size_t)(NH_ * HD_) * D_ / 8;
    cvt_f32_bf16_kernel<<<dim3((unsigned)((n8 + 255) / 256)), dim3(256), 0, stream>>>(Wq, Wqb, n8);
    n8 = (size_t)(NKV_ * HD_) * D_ / 8;
    cvt_f32_bf16_kernel<<<dim3((unsigned)((n8 + 255) / 256)), dim3(256), 0, stream>>>(Wk, Wkb, n8);
    cvt_f32_bf16_kernel<<<dim3((unsigned)((n8 + 255) / 256)), dim3(256), 0, stream>>>(Wv, Wvb, n8);
    n8 = (size_t)D_ * (NH_ * HD_) / 8;
    cvt_f32_bf16_kernel<<<dim3((unsigned)((n8 + 255) / 256)), dim3(256), 0, stream>>>(Wo, Wob, n8);
  }

  dim3 blk(128);
  // Q/K/V projections (x @ W^T), bf16 out; block tile 64x128
  gemm_bt_kernel<false><<<dim3(M / 64, (NH_ * HD_) / 128), blk, 0, stream>>>(
      Xb, Wqb, nullptr, Qb, M, NH_ * HD_, D_);
  gemm_bt_kernel<false><<<dim3(M / 64, (NKV_ * HD_) / 128), blk, 0, stream>>>(
      Xb, Wkb, nullptr, Kb, M, NKV_ * HD_, D_);
  gemm_bt_kernel<false><<<dim3(M / 64, (NKV_ * HD_) / 128), blk, 0, stream>>>(
      Xb, Wvb, nullptr, Vb, M, NKV_ * HD_, D_);

  // RoPE on Q and K (in place)
  {
    size_t nq = (size_t)B_ * S_ * NH_ * 32;
    rope_kernel<<<dim3((unsigned)((nq + 255) / 256)), dim3(256), 0, stream>>>(Qb, NH_);
    size_t nk = (size_t)B_ * S_ * NKV_ * 32;
    rope_kernel<<<dim3((unsigned)((nk + 255) / 256)), dim3(256), 0, stream>>>(Kb, NKV_);
  }

  // V transpose for contiguous P@V B-fragments
  {
    size_t nv = (size_t)B_ * S_ * NKV_ * HD_;
    vtrans_kernel<<<dim3((unsigned)((nv + 255) / 256)), dim3(256), 0, stream>>>(Vb, VTb);
  }

  // flash attention
  attn_kernel<<<dim3(S_ / 16, NH_, B_), dim3(32), 0, stream>>>(Qb, Kb, VTb, Ob);

  // output projection, fp32 out
  gemm_bt_kernel<true><<<dim3(M / 64, D_ / 128), blk, 0, stream>>>(
      Ob, Wob, out, nullptr, M, D_, NH_ * HD_);
}
